// TACDSR_13675175870766
// MI455X (gfx1250) — matrix operand order repaired and checked by reading, not compiled, not run on hardware
//
#include <hip/hip_runtime.h>
#include <hip/hip_bf16.h>
#include <math.h>

// ---------------- constants from the reference ----------------
#define ITEMNUM   100000
#define Hdim      128
#define NHh       2
#define NBb       2
#define Bb        128
#define Ll        200
#define NTOK      (Bb * Ll)          // 25600 tokens
#define NNZc      1600000
#define HD        64                 // head dim
#define FFH       512                // MLP hidden

typedef __attribute__((ext_vector_type(16))) _Float16 v16h;
typedef __attribute__((ext_vector_type(8)))  float    v8f;

// ---------------- small helpers ----------------
__device__ __forceinline__ float wave_sum(float v) {
#pragma unroll
  for (int o = 16; o > 0; o >>= 1) v += __shfl_xor(v, o, 32);
  return v;
}
__device__ __forceinline__ float wave_max(float v) {
#pragma unroll
  for (int o = 16; o > 0; o >>= 1) v = fmaxf(v, __shfl_xor(v, o, 32));
  return v;
}

// ------- weight pack: f32 row-major [S][K][N] -> f16 WMMA-B fragments -------
// packed index = ((s)*(K*N)) + (((nt*(K/32) + kc)*32 + lane)*16 + e)
// element (lane,e) of chunk (nt,kc) = W[kc*32 + (lane>>4)*16 + e][nt*16 + (lane&15)]
__global__ void pack_w_kernel(const float* __restrict__ src,
                              _Float16* __restrict__ dst,
                              int K, int N, int total) {
  int i = blockIdx.x * blockDim.x + threadIdx.x;
  if (i >= total) return;
  int per = K * N;
  int s = i / per, p = i % per;
  int e    = p & 15;
  int t    = p >> 4;
  int lane = t & 31;
  int t2   = t >> 5;
  int KC   = K >> 5;
  int kc   = t2 % KC;
  int nt   = t2 / KC;
  int k = kc * 32 + (lane >> 4) * 16 + e;
  int n = nt * 16 + (lane & 15);
  dst[i] = (_Float16)src[(size_t)s * per + (size_t)k * N + n];
}

// ---------------- SpMM scatter: acc[row] += val * emb[col] ----------------
__global__ void spmm_kernel(const int* __restrict__ rows,
                            const int* __restrict__ cols,
                            const float* __restrict__ val,
                            const float* __restrict__ emb,
                            float* acc, int nnz) {
  long long tid = (long long)blockIdx.x * blockDim.x + threadIdx.x;
  int nz = (int)(tid >> 5);
  if (nz >= nnz) return;
  int j = (int)(tid & 31);
  int r = rows[nz], c = cols[nz];
  float v = val[nz];
  const float* er = emb + (size_t)c * Hdim;
  float* ar = acc + (size_t)r * Hdim;
#pragma unroll
  for (int q = 0; q < 4; ++q) {
    int h = j + 32 * q;
    atomicAdd(&ar[h], v * er[h]);
  }
}

// --------- fused l2n + mix + gather: seq[t] = 1.5*emb[id] + 0.5*l2n(acc[id]) ---------
__global__ void gather_mix_kernel(const int* __restrict__ ids,
                                  const float* __restrict__ emb,
                                  const float* __restrict__ acc,
                                  float* __restrict__ seq, int ntok) {
  int wave = (blockIdx.x * blockDim.x + threadIdx.x) >> 5;
  int lane = threadIdx.x & 31;
  if (wave >= ntok) return;
  int id = ids[wave];
  const float* ar = acc + (size_t)id * Hdim;
  const float* er = emb + (size_t)id * Hdim;
  float av[4], ev[4], sq = 0.f;
#pragma unroll
  for (int j = 0; j < 4; ++j) {
    int h = lane + 32 * j;
    av[j] = ar[h]; ev[j] = er[h];
    sq += av[j] * av[j];
  }
  sq = wave_sum(sq);
  float inv = 1.f / fmaxf(sqrtf(sq), 1e-12f);
  float* sr = seq + (size_t)wave * Hdim;
#pragma unroll
  for (int j = 0; j < 4; ++j) {
    int h = lane + 32 * j;
    sr[h] = 1.5f * ev[j] + 0.5f * av[j] * inv;
  }
}

// --------- u[t] = interval_table[floor(log2(it+1))] + time_table[time[t]] ---------
__global__ void build_u_kernel(const int* __restrict__ tvals,
                               const int* __restrict__ ivals,
                               const float* __restrict__ time_table,
                               const float* __restrict__ interval_table,
                               float* __restrict__ u, int ntok) {
  int tid = blockIdx.x * blockDim.x + threadIdx.x;
  if (tid >= ntok * Hdim) return;
  int t = tid >> 7, h = tid & 127;
  int it = ivals[t];
  int idx = (int)floorf(log2f((float)it + 1.0f));
  u[tid] = interval_table[(size_t)idx * Hdim + h] +
           time_table[(size_t)tvals[t] * Hdim + h];
}

// ---------------- WMMA GEMM: C = epil(A(+A2) @ W + bias) ----------------
// one wave -> one 16x32 tile (two accumulators sharing the A fragment)
// K compile-time (multiple of 32); M mult of 16; N mult of 32
template <int K, bool HASA2, bool RELU, bool HASRES, bool HASKEEP>
__global__ __launch_bounds__(256) void gemm_wmma_kernel(
    const float* __restrict__ A, const float* __restrict__ A2,
    const _Float16* __restrict__ Wp, const float* __restrict__ bias,
    float* __restrict__ C, const float* __restrict__ residual,
    const float* __restrict__ keep, int M, int N) {
  constexpr int KC = K / 32;
  int wave = (int)((blockIdx.x * blockDim.x + threadIdx.x) >> 5);
  int lane = threadIdx.x & 31;
  int npair = N >> 5;
  int mt = wave / npair, np = wave % npair;
  if (mt >= (M >> 4)) return;
  int row0 = mt << 4, col0 = np << 5;
  int lrow = lane & 15, lhalf = lane >> 4, lcol = lane & 15;

  const float* arow  = A + (size_t)(row0 + lrow) * K + lhalf * 16;
  const float* arow2 = HASA2 ? A2 + (size_t)(row0 + lrow) * K + lhalf * 16 : nullptr;
  // per-lane packed-B bases: Wp + nt*KC*512 + lane*16  (stride 512 per kc)
  const _Float16* wp0 = Wp + (size_t)(2 * np)     * KC * 512 + lane * 16;
  const _Float16* wp1 = Wp + (size_t)(2 * np + 1) * KC * 512 + lane * 16;

  v8f acc0 = {}, acc1 = {};
#pragma unroll
  for (int kc = 0; kc < KC; ++kc) {
    const int kb = kc * 32;
    float4 p0 = *(const float4*)(arow + kb);
    float4 p1 = *(const float4*)(arow + kb + 4);
    float4 q0 = *(const float4*)(arow + kb + 8);
    float4 q1 = *(const float4*)(arow + kb + 12);
    if (HASA2) {
      float4 r0 = *(const float4*)(arow2 + kb);
      float4 r1 = *(const float4*)(arow2 + kb + 4);
      float4 s0 = *(const float4*)(arow2 + kb + 8);
      float4 s1 = *(const float4*)(arow2 + kb + 12);
      p0.x += r0.x; p0.y += r0.y; p0.z += r0.z; p0.w += r0.w;
      p1.x += r1.x; p1.y += r1.y; p1.z += r1.z; p1.w += r1.w;
      q0.x += s0.x; q0.y += s0.y; q0.z += s0.z; q0.w += s0.w;
      q1.x += s1.x; q1.y += s1.y; q1.z += s1.z; q1.w += s1.w;
    }
    v16h af;
    af[0]  = (_Float16)p0.x; af[1]  = (_Float16)p0.y;
    af[2]  = (_Float16)p0.z; af[3]  = (_Float16)p0.w;
    af[4]  = (_Float16)p1.x; af[5]  = (_Float16)p1.y;
    af[6]  = (_Float16)p1.z; af[7]  = (_Float16)p1.w;
    af[8]  = (_Float16)q0.x; af[9]  = (_Float16)q0.y;
    af[10] = (_Float16)q0.z; af[11] = (_Float16)q0.w;
    af[12] = (_Float16)q1.x; af[13] = (_Float16)q1.y;
    af[14] = (_Float16)q1.z; af[15] = (_Float16)q1.w;
    v16h bf0 = *(const v16h*)(wp0 + kc * 512);   // 2x global_load_b128
    v16h bf1 = *(const v16h*)(wp1 + kc * 512);
    acc0 = __builtin_amdgcn_wmma_f32_16x16x32_f16(false, af, false, bf0,
                                                  (short)0, acc0, false, false);
    acc1 = __builtin_amdgcn_wmma_f32_16x16x32_f16(false, af, false, bf1,
                                                  (short)0, acc1, false, false);
  }
  // C/D layout: vgpr r -> M = lhalf*8 + r, N = lane&15
  float kpv = 1.f;
#pragma unroll
  for (int r = 0; r < 8; ++r) {
    int m  = row0 + lhalf * 8 + r;
    if (HASKEEP && r == 0) kpv = keep[row0 + lhalf * 8];  // same m rows share? no:
    // keep depends on m; load per r (uniformized below)
    float km = HASKEEP ? keep[m] : 1.f;
    int n0 = col0 + lcol, n1 = col0 + 16 + lcol;
    float v0 = acc0[r] + bias[n0];
    float v1 = acc1[r] + bias[n1];
    if (RELU) { v0 = v0 > 0.f ? v0 : 0.f; v1 = v1 > 0.f ? v1 : 0.f; }
    if (HASRES) {
      v0 += residual[(size_t)m * N + n0];
      v1 += residual[(size_t)m * N + n1];
    }
    if (HASKEEP) { v0 *= km; v1 *= km; }
    C[(size_t)m * N + n0] = v0;
    C[(size_t)m * N + n1] = v1;
  }
  (void)kpv;
}

// ---------------- LayerNorm over H=128 (optionally on a+b) ----------------
__global__ void ln_kernel(const float* __restrict__ a, const float* b,
                          const float* __restrict__ g, const float* __restrict__ be,
                          float* __restrict__ out, int ntok, float eps) {
  int wave = (blockIdx.x * blockDim.x + threadIdx.x) >> 5;
  int lane = threadIdx.x & 31;
  if (wave >= ntok) return;
  const float* ar = a + (size_t)wave * Hdim;
  const float* br = b ? b + (size_t)wave * Hdim : nullptr;
  float x[4], s = 0.f;
#pragma unroll
  for (int j = 0; j < 4; ++j) {
    int h = lane + 32 * j;
    float v = ar[h];
    if (br) v += br[h];
    x[j] = v; s += v;
  }
  float mu = wave_sum(s) * (1.f / Hdim);
  float vs = 0.f;
#pragma unroll
  for (int j = 0; j < 4; ++j) { float d = x[j] - mu; vs += d * d; }
  float var = wave_sum(vs) * (1.f / Hdim);
  float inv = rsqrtf(var + eps);
  float* orow = out + (size_t)wave * Hdim;
#pragma unroll
  for (int j = 0; j < 4; ++j) {
    int h = lane + 32 * j;
    orow[h] = (x[j] - mu) * inv * g[h] + be[h];
  }
}

// ------- encoder pre: seq=(seq+p)*keep; temb+=p; gates=sigmoid(temb@gW+gb) -------
__global__ void encoder_pre_kernel(const int* __restrict__ ids,
                                   const int* __restrict__ pos,
                                   const float* __restrict__ pemb,
                                   const float* __restrict__ gW,
                                   const float* __restrict__ gb,
                                   float* seq, float* temb,
                                   float* gates, float* keep, int ntok) {
  int wave = (blockIdx.x * blockDim.x + threadIdx.x) >> 5;
  int lane = threadIdx.x & 31;
  if (wave >= ntok) return;
  int id = ids[wave];
  float kp = (id != ITEMNUM - 1) ? 1.f : 0.f;
  const float* pr = pemb + (size_t)pos[wave] * Hdim;
  float* sr = seq + (size_t)wave * Hdim;
  float* tr = temb + (size_t)wave * Hdim;
  float g0 = 0.f, g1 = 0.f;
#pragma unroll
  for (int j = 0; j < 4; ++j) {
    int h = lane + 32 * j;
    float pv = pr[h];
    float sv = (sr[h] + pv) * kp;
    float tv = tr[h] + pv;
    sr[h] = sv; tr[h] = tv;
    g0 += tv * gW[h * NBb + 0];
    g1 += tv * gW[h * NBb + 1];
  }
  g0 = wave_sum(g0); g1 = wave_sum(g1);
  if (lane == 0) {
    keep[wave] = kp;
    gates[(size_t)wave * NBb + 0] = 1.f / (1.f + __expf(-(g0 + gb[0])));
    gates[(size_t)wave * NBb + 1] = 1.f / (1.f + __expf(-(g1 + gb[1])));
  }
}

// ------- gated causal attention: one wave per (b, h, q) row -------
__global__ __launch_bounds__(256) void attn_kernel(
    const float* __restrict__ Qm, const float* __restrict__ Km,
    const float* __restrict__ Vm, const float* __restrict__ gates,
    float* __restrict__ Om, int gi) {
  __shared__ float qsh[8][HD];
  __shared__ float pr[8][Ll];
  int wslot = threadIdx.x >> 5, lane = threadIdx.x & 31;
  int wid = blockIdx.x * 8 + wslot;               // 0 .. 51199 (exact)
  int b   = wid / (NHh * Ll);
  int rem = wid % (NHh * Ll);
  int h   = rem / Ll;
  int q   = rem % Ll;

  const float* qrow = Qm + (size_t)(b * Ll + q) * Hdim + h * HD;
  qsh[wslot][lane]      = qrow[lane];
  qsh[wslot][lane + 32] = qrow[lane + 32];
  float gq = gates[(size_t)(b * Ll + q) * NBb + gi];

  // pass 1: raw gated scores for keys 0..q (causal)
  float lmax = -1e30f;
  for (int kk = lane; kk <= q; kk += 32) {
    const float* krow = Km + (size_t)(b * Ll + kk) * Hdim + h * HD;
    float dot = 0.f;
#pragma unroll 8
    for (int d = 0; d < HD; ++d) dot += qsh[wslot][d] * krow[d];
    float gk = gates[(size_t)(b * Ll + kk) * NBb + gi];
    float sc = dot * 0.125f * gq * gk;            // scale = 1/sqrt(64)
    pr[wslot][kk] = sc;
    lmax = fmaxf(lmax, sc);
  }
  lmax = wave_max(lmax);
  float lsum = 0.f;
  for (int kk = lane; kk <= q; kk += 32) {
    float p = __expf(pr[wslot][kk] - lmax);
    pr[wslot][kk] = p;
    lsum += p;
  }
  lsum = wave_sum(lsum);
  float inv = 1.f / lsum;

  // pass 2: weighted V; each lane owns dims lane, lane+32
  float a0 = 0.f, a1 = 0.f;
  for (int kk = 0; kk <= q; ++kk) {
    float p = pr[wslot][kk] * inv;
    const float* vrow = Vm + (size_t)(b * Ll + kk) * Hdim + h * HD;
    a0 += p * vrow[lane];
    a1 += p * vrow[lane + 32];
  }
  float* orow = Om + (size_t)(b * Ll + q) * Hdim + h * HD;
  orow[lane] = a0; orow[lane + 32] = a1;
}

// ======================= host launcher =======================
extern "C" void kernel_launch(void* const* d_in, const int* in_sizes, int n_in,
                              void* d_out, int out_size, void* d_ws, size_t ws_size,
                              hipStream_t stream) {
  // -------- inputs (setup_inputs dict order) --------
  const int*   o_seqs      = (const int*)d_in[0];
  const int*   x_seqs      = (const int*)d_in[1];
  const int*   y_seqs      = (const int*)d_in[2];
  const int*   position    = (const int*)d_in[3];
  const int*   x_position  = (const int*)d_in[4];
  const int*   y_position  = (const int*)d_in[5];
  const int*   time_       = (const int*)d_in[6];
  const int*   x_time      = (const int*)d_in[7];
  const int*   y_time      = (const int*)d_in[8];
  const int*   x_itime     = (const int*)d_in[9];
  const int*   y_itime     = (const int*)d_in[10];
  const int*   itime       = (const int*)d_in[11];
  const float* embX        = (const float*)d_in[12];
  const float* embY        = (const float*)d_in[13];
  const float* embO        = (const float*)d_in[14];
  const float* time_table  = (const float*)d_in[15];
  const float* interval_table = (const float*)d_in[16];
  const int*   adj_idx     = (const int*)d_in[17];
  const float* adj_val     = (const float*)d_in[18];
  const int*   adjs_idx    = (const int*)d_in[19];
  const float* adjs_val    = (const float*)d_in[20];
  const float* mlp_W1      = (const float*)d_in[21];
  const float* mlp_b1      = (const float*)d_in[22];
  const float* mlp_W2      = (const float*)d_in[23];
  const float* mlp_b2      = (const float*)d_in[24];
  const float* mlp_lng     = (const float*)d_in[25];
  const float* mlp_lnb     = (const float*)d_in[26];
  const float* pos_emb     = (const float*)d_in[27];
  const float* gate_W      = (const float*)d_in[28];
  const float* gate_b      = (const float*)d_in[29];
  const float* aln_g       = (const float*)d_in[30];
  const float* aln_b       = (const float*)d_in[31];
  const float* qW          = (const float*)d_in[32];
  const float* kW          = (const float*)d_in[33];
  const float* vW          = (const float*)d_in[34];
  const float* c1W         = (const float*)d_in[35];
  const float* c2W         = (const float*)d_in[36];
  const float* qb          = (const float*)d_in[37];
  const float* kb          = (const float*)d_in[38];
  const float* vb          = (const float*)d_in[39];
  const float* c1b         = (const float*)d_in[40];
  const float* c2b         = (const float*)d_in[41];
  const float* fln_g       = (const float*)d_in[42];
  const float* fln_b       = (const float*)d_in[43];
  const float* lln_g       = (const float*)d_in[44];
  const float* lln_b       = (const float*)d_in[45];

  // -------- workspace layout (floats) --------
  const size_t MH  = (size_t)NTOK * Hdim;     // 3,276,800
  const size_t M5  = (size_t)NTOK * FFH;      // 13,107,200 (also SpMM acc: 100000*128 fits)
  float* ws   = (float*)d_ws;
  float* FA   = ws;                 // [M5]  h512 scratch / spmm accumulator
  float* seqb = FA   + M5;          // [MH]
  float* temb = seqb + MH;          // [MH]
  float* t1   = temb + MH;          // [MH]  Q / mlp-input u
  float* t2   = t1   + MH;          // [MH]  q / ffn hidden
  float* t3   = t2   + MH;          // [MH]  k
  float* t4   = t3   + MH;          // [MH]  v
  float* t5   = t4   + MH;          // [MH]  attn out
  float* gatesb = t5 + MH;          // [NTOK*2]
  float* keepb  = gatesb + (size_t)NTOK * 2;  // [NTOK]
  size_t fcount = (keepb + NTOK) - ws;
  _Float16* hbase = (_Float16*)(ws + fcount);
  _Float16* W1h = hbase;                       // 3*128*512 (packed)
  _Float16* W2h = W1h + 3 * 128 * 512;         // 3*512*128 (packed)
  _Float16* qWh = W2h + 3 * 512 * 128;         // 3*2*128*128 each (packed)
  _Float16* kWh = qWh + 6 * 128 * 128;
  _Float16* vWh = kWh + 6 * 128 * 128;
  _Float16* c1Wh = vWh + 6 * 128 * 128;
  _Float16* c2Wh = c1Wh + 6 * 128 * 128;

  // -------- pack weights to f16 WMMA-B fragment layout (every call) --------
  auto pack = [&](const float* s, _Float16* d, int S, int K, int N) {
    int total = S * K * N;
    pack_w_kernel<<<(total + 255) / 256, 256, 0, stream>>>(s, d, K, N, total);
  };
  pack(mlp_W1, W1h, 3, 128, 512);
  pack(mlp_W2, W2h, 3, 512, 128);
  pack(qW,  qWh,  6, 128, 128);
  pack(kW,  kWh,  6, 128, 128);
  pack(vW,  vWh,  6, 128, 128);
  pack(c1W, c1Wh, 6, 128, 128);
  pack(c2W, c2Wh, 6, 128, 128);

  // -------- GEMM dispatch helpers (template instantiations) --------
  // waves = (M/16)*(N/32); blocks = waves/8 (exact for our shapes)
  auto g_mlp1 = [&](const float* A, const _Float16* W, const float* bias, float* C) {
    gemm_wmma_kernel<128, false, true, false, false>
        <<<(NTOK / 16) * (FFH / 32) / 8, 256, 0, stream>>>(
            A, nullptr, W, bias, C, nullptr, nullptr, NTOK, FFH);
  };
  auto g_mlp2 = [&](const float* A, const _Float16* W, const float* bias, float* C) {
    gemm_wmma_kernel<512, false, false, false, false>
        <<<(NTOK / 16) * (Hdim / 32) / 8, 256, 0, stream>>>(
            A, nullptr, W, bias, C, nullptr, nullptr, NTOK, Hdim);
  };
  auto g_proj = [&](const float* A, const _Float16* W, const float* bias, float* C) {
    gemm_wmma_kernel<128, false, false, false, false>
        <<<(NTOK / 16) * (Hdim / 32) / 8, 256, 0, stream>>>(
            A, nullptr, W, bias, C, nullptr, nullptr, NTOK, Hdim);
  };
  auto g_proj2 = [&](const float* A, const float* A2, const _Float16* W,
                     const float* bias, float* C) {
    gemm_wmma_kernel<128, true, false, false, false>
        <<<(NTOK / 16) * (Hdim / 32) / 8, 256, 0, stream>>>(
            A, A2, W, bias, C, nullptr, nullptr, NTOK, Hdim);
  };
  auto g_ffn1 = [&](const float* A, const _Float16* W, const float* bias, float* C) {
    gemm_wmma_kernel<128, false, true, false, false>
        <<<(NTOK / 16) * (Hdim / 32) / 8, 256, 0, stream>>>(
            A, nullptr, W, bias, C, nullptr, nullptr, NTOK, Hdim);
  };
  auto g_ffn2 = [&](const float* A, const _Float16* W, const float* bias, float* C,
                    const float* res, const float* keep) {
    gemm_wmma_kernel<128, false, false, true, true>
        <<<(NTOK / 16) * (Hdim / 32) / 8, 256, 0, stream>>>(
            A, nullptr, W, bias, C, res, keep, NTOK, Hdim);
  };
  auto lnorm = [&](const float* a, const float* b, const float* g,
                   const float* be, float* out, float eps) {
    ln_kernel<<<NTOK / 8, 256, 0, stream>>>(a, b, g, be, out, NTOK, eps);
  };

  // -------- per-encoder routing --------
  const int*   ids_arr[3]  = {o_seqs, x_seqs, y_seqs};
  const int*   pos_arr[3]  = {position, x_position, y_position};
  const int*   tm_arr[3]   = {time_, x_time, y_time};
  const int*   it_arr[3]   = {itime, x_itime, y_itime};
  const float* emb_arr[3]  = {embO, embX, embY};
  const int*   aidx_arr[3] = {adj_idx, adjs_idx, adjs_idx};
  const float* aval_arr[3] = {adj_val, adjs_val, adjs_val};
  const int    mlp_idx[3]  = {2, 0, 1};

  for (int e = 0; e < 3; ++e) {
    const int m = mlp_idx[e];

    // --- SpMM + mix + gather: seqb[t] = 1.5*emb[id] + 0.5*l2n(spmm)[id] ---
    hipMemsetAsync(FA, 0, (size_t)ITEMNUM * Hdim * sizeof(float), stream);
    {
      long long thr = (long long)NNZc * 32;
      spmm_kernel<<<(unsigned)((thr + 255) / 256), 256, 0, stream>>>(
          aidx_arr[e], aidx_arr[e] + NNZc, aval_arr[e], emb_arr[e], FA, NNZc);
    }
    gather_mix_kernel<<<NTOK / 8, 256, 0, stream>>>(ids_arr[e], emb_arr[e], FA,
                                                    seqb, NTOK);

    // --- temb = LN(relu(u@W1+b1)@W2+b2) ---
    build_u_kernel<<<(NTOK * Hdim) / 256, 256, 0, stream>>>(
        tm_arr[e], it_arr[e], time_table, interval_table, t1, NTOK);
    g_mlp1(t1, W1h + (size_t)m * 128 * 512, mlp_b1 + m * 512, FA);
    g_mlp2(FA, W2h + (size_t)m * 512 * 128, mlp_b2 + m * 128, temb);
    lnorm(temb, nullptr, mlp_lng + m * 128, mlp_lnb + m * 128, temb, 1e-5f);

    // --- pre: add pos emb, keep mask, gates ---
    encoder_pre_kernel<<<NTOK / 8, 256, 0, stream>>>(
        ids_arr[e], pos_arr[e], pos_emb + (size_t)e * Ll * Hdim,
        gate_W + (size_t)e * Hdim * NBb, gate_b + (size_t)e * NBb, seqb, temb,
        gatesb, keepb, NTOK);

    // --- transformer blocks ---
    for (int i = 0; i < NBb; ++i) {
      const size_t ei = (size_t)e * NBb + i;
      // Q = LN(seq)
      lnorm(seqb, nullptr, aln_g + ei * Hdim, aln_b + ei * Hdim, t1, 1e-8f);
      // q/k/v projections (WMMA)
      g_proj (t1,         qWh + ei * Hdim * Hdim, qb + ei * Hdim, t2);
      g_proj (temb,       kWh + ei * Hdim * Hdim, kb + ei * Hdim, t3);
      g_proj2(seqb, temb, vWh + ei * Hdim * Hdim, vb + ei * Hdim, t4);
      // gated causal attention
      attn_kernel<<<(Bb * NHh * Ll) / 8, 256, 0, stream>>>(t2, t3, t4, gatesb,
                                                           t5, i);
      // seq = LN(Q + attn_out)
      lnorm(t1, t5, fln_g + ei * Hdim, fln_b + ei * Hdim, seqb, 1e-8f);
      // FFN: seq = (seq + relu(seq@c1+b1)@c2 + b2) * keep
      g_ffn1(seqb, c1Wh + ei * Hdim * Hdim, c1b + ei * Hdim, t2);
      g_ffn2(t2, c2Wh + ei * Hdim * Hdim, c2b + ei * Hdim, seqb, seqb, keepb);
    }
    // --- final LN -> output ---
    lnorm(seqb, nullptr, lln_g + (size_t)e * Hdim, lln_b + (size_t)e * Hdim,
          (float*)d_out + (size_t)e * MH, 1e-8f);
  }
}
